// FeaturePropagation_28097676051192
// MI455X (gfx1250) — matrix operand order repaired and checked by reading, not compiled, not run on hardware
//
#include <hip/hip_runtime.h>
#include <hip/hip_bf16.h>

typedef unsigned short u16;
typedef unsigned int   u32;
typedef unsigned long long u64;

typedef __attribute__((ext_vector_type(16))) __bf16 v16bf;
typedef __attribute__((ext_vector_type(8)))  float  v8f;

// ---------------------------------------------------------------- constants
#define BB   4
#define NN   16384
#define SS   2048
#define C1C  128
#define C2C  256
#define CIN  384      // C1 + C2
#define CH   256
#define MTOT (BB * NN)   // 65536 rows

// ---------------------------------------------------------------- helpers
__device__ __forceinline__ u16 f2bf(float x) {
    u32 u = __float_as_uint(x);
    u32 r = (u + 0x7fffu + ((u >> 16) & 1u)) >> 16;   // round-to-nearest-even
    return (u16)r;
}
__device__ __forceinline__ u32 pack_bf2(float lo, float hi) {
    return (u32)f2bf(lo) | ((u32)f2bf(hi) << 16);
}

// CDNA5 async global->LDS copy (ASYNCcnt-tracked, bypasses VGPRs).
// lds_addr: LDS byte address (low 32 bits of the generic pointer), per lane.
__device__ __forceinline__ void async_load_b128(u32 lds_addr, const void* gaddr) {
    asm volatile("global_load_async_to_lds_b128 %0, %1, off"
                 :: "v"(lds_addr), "v"(gaddr)
                 : "memory");
}
__device__ __forceinline__ void wait_async0() {
    asm volatile("s_wait_asynccnt 0x0" ::: "memory");
}

// ---------------------------------------------------------------- fp32 -> bf16 copy
__global__ __launch_bounds__(256)
void f32_to_bf16_kernel(const float* __restrict__ in, u16* __restrict__ out, int n) {
    int i = blockIdx.x * 256 + threadIdx.x;
    if (i < n) out[i] = f2bf(in[i]);
}

// ---------------------------------------------------------------- KNN (K=3) + inverse-distance interp
// Writes concatenated bf16 feature rows: [feat1 (128) | interp (256)] per point.
__global__ __launch_bounds__(256)
void knn_interp_kernel(const float* __restrict__ p1, const float* __restrict__ p2,
                       const float* __restrict__ f1, const float* __restrict__ f2,
                       u16* __restrict__ out) {
    __shared__ float sx[SS], sy[SS], sz[SS];
    const int b   = blockIdx.y;
    const int tid = threadIdx.x;

    for (int i = tid; i < SS; i += 256) {
        const float* q = p2 + ((size_t)b * SS + i) * 3;
        sx[i] = q[0]; sy[i] = q[1]; sz[i] = q[2];
    }
    __syncthreads();

    const int n = blockIdx.x * 256 + tid;
    const float* p = p1 + ((size_t)b * NN + n) * 3;
    const float px = p[0], py = p[1], pz = p[2];

    float d0 = 3.4e38f, d1 = 3.4e38f, d2 = 3.4e38f;
    int   i0 = 0, i1 = 0, i2 = 0;
    for (int s = 0; s < SS; ++s) {
        float dx = px - sx[s], dy = py - sy[s], dz = pz - sz[s];
        float d  = dx * dx + dy * dy + dz * dz;
        if (d < d2) {
            if (d < d1) {
                d2 = d1; i2 = i1;
                if (d < d0) { d1 = d0; i1 = i0; d0 = d; i0 = s; }
                else        { d1 = d;  i1 = s; }
            } else { d2 = d; i2 = s; }
        }
    }

    float w0 = 1.0f / (d0 + 1e-8f);
    float w1 = 1.0f / (d1 + 1e-8f);
    float w2 = 1.0f / (d2 + 1e-8f);
    const float inv = 1.0f / (w0 + w1 + w2);
    w0 *= inv; w1 *= inv; w2 *= inv;

    u16* dst = out + ((size_t)b * NN + n) * CIN;

    // feat1 -> bf16
    const float* s1 = f1 + ((size_t)b * NN + n) * C1C;
    u32* d32 = (u32*)dst;
    #pragma unroll 4
    for (int c = 0; c < C1C; c += 2)
        d32[c >> 1] = pack_bf2(s1[c], s1[c + 1]);

    // interpolated feat2 -> bf16
    const float* r0 = f2 + ((size_t)b * SS + i0) * C2C;
    const float* r1 = f2 + ((size_t)b * SS + i1) * C2C;
    const float* r2 = f2 + ((size_t)b * SS + i2) * C2C;
    u32* di = (u32*)(dst + C1C);
    #pragma unroll 4
    for (int c = 0; c < C2C; c += 2) {
        float v0 = w0 * r0[c]     + w1 * r1[c]     + w2 * r2[c];
        float v1 = w0 * r0[c + 1] + w1 * r1[c + 1] + w2 * r2[c + 1];
        di[c >> 1] = pack_bf2(v0, v1);
    }
}

// ---------------------------------------------------------------- bf16 WMMA GEMM
// C[M,256] = A[M,K](bf16) * Bw[256,K](bf16, row = output channel), f32 accumulate.
// Block: 256 threads = 8 waves; block tile 64(M) x 256(N); k-step 32.
// Double-buffered LDS staged via global_load_async_to_lds_b128 (ASYNCcnt).
// Wave w: row-tile rt = w>>1 (16 rows), col half chf = w&1 (8 col tiles of 16).
__global__ __launch_bounds__(256)
void gemm_bf16_kernel(const u16* __restrict__ A, const u16* __restrict__ Bw,
                      float* __restrict__ C, int K) {
    __shared__ u16 sA[2][64 * 32];    //  2 x  4 KB  [row][k]
    __shared__ u16 sB[2][256 * 32];   //  2 x 16 KB  [n][k]

    const int tid  = threadIdx.x;
    const int wave = tid >> 5;
    const int lane = tid & 31;
    const int rt   = wave >> 1;
    const int chf  = wave & 1;
    const int hi   = lane >> 4;       // lane group (0: lanes 0-15, 1: lanes 16-31)
    const int l15  = lane & 15;
    const int rowBase = blockIdx.x * 64;

    // per-thread staging source/dest (k0 added per step)
    const int ar = tid >> 2;
    const int ac = (tid & 3) * 8;
    const u16* gA = A  + (size_t)(rowBase + ar) * K + ac;
    const u16* gB = Bw + (size_t)tid * K;
    const u32 ldsA0 = (u32)(u64)(uintptr_t)&sA[0][ar * 32 + ac];
    const u32 ldsB0 = (u32)(u64)(uintptr_t)&sB[0][tid * 32];
    const u32 strideA = 64 * 32 * 2;     // bytes per A buffer
    const u32 strideB = 256 * 32 * 2;    // bytes per B buffer

    v8f acc[8];
    #pragma unroll
    for (int t = 0; t < 8; ++t) acc[t] = (v8f){0.f, 0.f, 0.f, 0.f, 0.f, 0.f, 0.f, 0.f};

    // prologue: stage k-step 0 into buffer 0
    {
        async_load_b128(ldsA0, gA);
        const u32 lb = ldsB0;
        async_load_b128(lb,      gB);
        async_load_b128(lb + 16, gB + 8);
        async_load_b128(lb + 32, gB + 16);
        async_load_b128(lb + 48, gB + 24);
    }
    wait_async0();
    __syncthreads();

    const int nk = K >> 5;
    for (int ks = 0; ks < nk; ++ks) {
        const int buf = ks & 1;

        // prefetch next k-step into the other buffer (async, overlaps WMMA below)
        if (ks + 1 < nk) {
            const int k0n = (ks + 1) << 5;
            const u32 nb = (buf ^ 1);
            async_load_b128(ldsA0 + nb * strideA, gA + k0n);
            const u32 lb = ldsB0 + nb * strideB;
            const u16* gb = gB + k0n;
            async_load_b128(lb,      gb);
            async_load_b128(lb + 16, gb + 8);
            async_load_b128(lb + 32, gb + 16);
            async_load_b128(lb + 48, gb + 24);
        }

        // A fragment (16x32 bf16 layout): lane row = l15;
        // halves 0-15 -> K = hi*16 + 0..15
        union { uint4 u[2]; v16bf v; } af;
        {
            const u16* base = &sA[buf][(rt * 16 + l15) * 32];
            af.u[0] = *(const uint4*)(base + hi * 16);
            af.u[1] = *(const uint4*)(base + hi * 16 + 8);
        }

        #pragma unroll
        for (int t = 0; t < 8; ++t) {
            // B fragment (32x16 bf16 layout): lane col = l15; halves h -> K = hi*16 + h
            union { uint4 u[2]; v16bf v; } bfm;
            const u16* base = &sB[buf][(chf * 128 + t * 16 + l15) * 32 + hi * 16];
            bfm.u[0] = ((const uint4*)base)[0];
            bfm.u[1] = ((const uint4*)base)[1];
            acc[t] = __builtin_amdgcn_wmma_f32_16x16x32_bf16(
                false, af.v, false, bfm.v, (short)0, acc[t], false, false);
        }

        // next-buffer async writes complete + all waves done reading this buffer
        wait_async0();
        __syncthreads();
    }

    // D layout: VGPR r -> M = rt*16 + hi*8 + r ; lane col = l15
    #pragma unroll
    for (int t = 0; t < 8; ++t) {
        const int col = chf * 128 + t * 16 + l15;
        #pragma unroll
        for (int r = 0; r < 8; ++r) {
            const int row = rowBase + rt * 16 + hi * 8 + r;
            C[(size_t)row * CH + col] = acc[t][r];
        }
    }
}

// ---------------------------------------------------------------- deterministic per-channel stats
// Phase 1: block j reduces rows [j*256, j*256+256); thread t owns channel t.
__global__ __launch_bounds__(256)
void col_stats_partial_kernel(const float* __restrict__ y, float* __restrict__ part) {
    const int j = blockIdx.x;
    const int t = threadIdx.x;
    const float* p = y + (size_t)j * 256 * CH + t;
    float s = 0.f, q = 0.f;
    for (int r = 0; r < 256; ++r) {
        float v = p[(size_t)r * CH];
        s += v; q += v * v;
    }
    part[j * 512 + t]       = s;
    part[j * 512 + 256 + t] = q;
}

// Phase 2: fixed-order reduction over the 256 partials -> stats[0..255]=sum, [256..511]=sumsq
__global__ __launch_bounds__(256)
void col_stats_final_kernel(const float* __restrict__ part, float* __restrict__ stats) {
    const int t = threadIdx.x;
    float s = 0.f, q = 0.f;
    for (int j = 0; j < 256; ++j) {
        s += part[j * 512 + t];
        q += part[j * 512 + 256 + t];
    }
    stats[t]       = s;
    stats[256 + t] = q;
}

// ---------------------------------------------------------------- BN(train) + ReLU
__global__ __launch_bounds__(256)
void bn_relu_bf16_kernel(const float* __restrict__ y, const float* __restrict__ stats,
                         const float* __restrict__ gamma, const float* __restrict__ beta,
                         u16* __restrict__ xbf) {
    const size_t i = (size_t)blockIdx.x * 256 + threadIdx.x;
    const int c = (int)(i & (CH - 1));
    const float invM = 1.0f / (float)MTOT;
    const float mean = stats[c] * invM;
    const float var  = stats[CH + c] * invM - mean * mean;
    float v = (y[i] - mean) * rsqrtf(var + 1e-5f) * gamma[c] + beta[c];
    xbf[i] = f2bf(fmaxf(v, 0.0f));
}

__global__ __launch_bounds__(256)
void bn_relu_f32_kernel(const float* __restrict__ y, const float* __restrict__ stats,
                        const float* __restrict__ gamma, const float* __restrict__ beta,
                        float* __restrict__ outp) {
    const size_t i = (size_t)blockIdx.x * 256 + threadIdx.x;
    const int c = (int)(i & (CH - 1));
    const float invM = 1.0f / (float)MTOT;
    const float mean = stats[c] * invM;
    const float var  = stats[CH + c] * invM - mean * mean;
    float v = (y[i] - mean) * rsqrtf(var + 1e-5f) * gamma[c] + beta[c];
    outp[i] = fmaxf(v, 0.0f);
}

// ---------------------------------------------------------------- launch
extern "C" void kernel_launch(void* const* d_in, const int* in_sizes, int n_in,
                              void* d_out, int out_size, void* d_ws, size_t ws_size,
                              hipStream_t stream) {
    const float* point_1      = (const float*)d_in[0];
    const float* point_2      = (const float*)d_in[1];
    const float* point_feat_1 = (const float*)d_in[2];
    const float* point_feat_2 = (const float*)d_in[3];
    const float* W1           = (const float*)d_in[4];
    const float* gamma1       = (const float*)d_in[5];
    const float* beta1        = (const float*)d_in[6];
    const float* W2           = (const float*)d_in[7];
    const float* gamma2       = (const float*)d_in[8];
    const float* beta2        = (const float*)d_in[9];
    float* out = (float*)d_out;

    // workspace layout (A1 region reused for x1 bf16; y2 reuses y1 buffer)
    char* ws = (char*)d_ws;
    const size_t A1_BYTES = (size_t)MTOT * CIN * 2;   // 50,331,648
    const size_t Y_BYTES  = (size_t)MTOT * CH * 4;    // 67,108,864
    u16*   A1    = (u16*)ws;                          // [M,384] bf16 (later x1 [M,256] bf16)
    float* ybuf  = (float*)(ws + A1_BYTES);           // [M,256] f32 (y1, then y2)
    u16*   W1b   = (u16*)(ws + A1_BYTES + Y_BYTES);            // [256,384] bf16
    u16*   W2b   = (u16*)((char*)W1b + (size_t)CH * CIN * 2);  // [256,256] bf16
    float* part  = (float*)((char*)W2b + (size_t)CH * CH * 2); // 256*512 f32
    float* st1   = (float*)((char*)part + 256 * 512 * 4);      // 512 f32
    float* st2   = st1 + 512;                                  // 512 f32
    u16*   x1bf  = A1;                                         // alias (A1 dead after GEMM1)

    // 1) weights -> bf16 (kept as [out_ch][in_ch] == WMMA B [N][K])
    f32_to_bf16_kernel<<<(CH * CIN + 255) / 256, 256, 0, stream>>>(W1, W1b, CH * CIN);
    f32_to_bf16_kernel<<<(CH * CH  + 255) / 256, 256, 0, stream>>>(W2, W2b, CH * CH);

    // 2) KNN + interpolation -> concatenated bf16 features A1 [M,384]
    knn_interp_kernel<<<dim3(NN / 256, BB), 256, 0, stream>>>(
        point_1, point_2, point_feat_1, point_feat_2, A1);

    // 3) GEMM1: y1 = A1 * W1^T  (f32 accum)
    gemm_bf16_kernel<<<MTOT / 64, 256, 0, stream>>>(A1, W1b, ybuf, CIN);

    // 4) BN1 stats (deterministic two-phase)
    col_stats_partial_kernel<<<256, 256, 0, stream>>>(ybuf, part);
    col_stats_final_kernel<<<1, 256, 0, stream>>>(part, st1);

    // 5) x1 = bf16(relu(bn1(y1)))
    bn_relu_bf16_kernel<<<MTOT, 256, 0, stream>>>(ybuf, st1, gamma1, beta1, x1bf);

    // 6) GEMM2: y2 = x1 * W2^T (overwrites y1 buffer)
    gemm_bf16_kernel<<<MTOT / 64, 256, 0, stream>>>(x1bf, W2b, ybuf, CH);

    // 7) BN2 stats
    col_stats_partial_kernel<<<256, 256, 0, stream>>>(ybuf, part);
    col_stats_final_kernel<<<1, 256, 0, stream>>>(part, st2);

    // 8) out = relu(bn2(y2)) fp32
    bn_relu_f32_kernel<<<MTOT, 256, 0, stream>>>(ybuf, st2, gamma2, beta2, out);
}
